// mini_encoder2_41532333752450
// MI455X (gfx1250) — matrix operand order repaired and checked by reading, not compiled, not run on hardware
//
#include <hip/hip_runtime.h>

typedef __bf16 bf16_t;
typedef __attribute__((ext_vector_type(16))) __bf16 v16bf;
typedef __attribute__((ext_vector_type(8)))  float  v8f;

#define BN_EPS 1e-5f

// ---------------------------------------------------------------------------
// WMMA helpers (CDNA5, wave32): D = A(16x32 bf16) x B(32x16 bf16) + C(f32)
// ---------------------------------------------------------------------------
__device__ __forceinline__ v8f wmma_bf16(v16bf a, v16bf b, v8f c) {
  return __builtin_amdgcn_wmma_f32_16x16x32_bf16(false, a, false, b, (short)0, c,
                                                 false, false);
}

// A-matrix 16x32 bf16 tile from row-major [row][k] storage.
// Layout used here (contiguous k, matched by B): lanes 0-15 = M rows; VGPR v, pos p -> K = half*16+v*2+p
__device__ __forceinline__ v16bf load_a_tile(const bf16_t* __restrict__ base,
                                             int ld, int row0, int k0) {
  const int lane = threadIdx.x & 31;
  const bf16_t* p = base + (size_t)(row0 + (lane & 15)) * ld + k0 + ((lane >> 4) << 4);
  v16bf t;
#pragma unroll
  for (int v = 0; v < 8; ++v) {
    const int k = v << 1;
    t[2 * v]     = p[k];
    t[2 * v + 1] = p[k + 1];
  }
  return t;
}

// B-matrix 32x16 bf16 tile; weights stored row-major as W'[n][k] (n = c_out).
// Layout used here (contiguous k, matched by A): lane = col n (dup halves); half 0 holds K=k0..k0+15, half 1 K=k0+16..k0+31.
__device__ __forceinline__ v16bf load_b_tile(const bf16_t* __restrict__ base,
                                             int ld, int n0, int k0) {
  const int lane = threadIdx.x & 31;
  const bf16_t* p = base + (size_t)(n0 + (lane & 15)) * ld + k0 + ((lane >> 4) << 4);
  v16bf t;
#pragma unroll
  for (int v = 0; v < 8; ++v) {
    t[2 * v]     = p[2 * v];
    t[2 * v + 1] = p[2 * v + 1];
  }
  return t;
}

// Store D tile (16x16 f32) with bias+ReLU as bf16 activation [m][k] for next layer.
// D layout: lane = col n (n = lane&15), VGPR v holds M = v + 8*(lane>>4).
__device__ __forceinline__ void store_act_tile(bf16_t* __restrict__ dst, int ld,
                                               int row0, int n0, v8f acc,
                                               const float* __restrict__ bias) {
  const int lane = threadIdx.x & 31;
  const int n = n0 + (lane & 15);
  const float bn = bias[n];
  bf16_t* p = dst + (size_t)(row0 + ((lane >> 4) << 3)) * ld + n;
#pragma unroll
  for (int v = 0; v < 8; ++v) {
    float x = acc[v] + bn;
    p[v * ld] = (bf16_t)(x > 0.f ? x : 0.f);
  }
}

#define V8F_ZERO {0.f, 0.f, 0.f, 0.f, 0.f, 0.f, 0.f, 0.f}

// ---------------------------------------------------------------------------
// Fold BN + bias into bf16 weights: W'[o][k] = W[o][k]*s, s = gamma*rsqrt(var+eps)
//                                   b'[o]    = (b[o]-mean[o])*s + beta[o]
// ---------------------------------------------------------------------------
__global__ void fold_kernel(const float* __restrict__ W, const float* __restrict__ bb,
                            const float* __restrict__ gamma, const float* __restrict__ beta,
                            const float* __restrict__ mean, const float* __restrict__ var,
                            bf16_t* __restrict__ Wout, float* __restrict__ bout,
                            int c_in, int k_pad) {
  const int o = blockIdx.x;
  const float s = gamma[o] * rsqrtf(var[o] + BN_EPS);
  for (int k = threadIdx.x; k < k_pad; k += blockDim.x)
    Wout[(size_t)o * k_pad + k] = (k < c_in) ? (bf16_t)(W[(size_t)o * c_in + k] * s)
                                             : (bf16_t)0.f;
  if (threadIdx.x == 0) bout[o] = (bb[o] - mean[o]) * s + beta[o];
}

// ---------------------------------------------------------------------------
// Farthest point sampling: one block per batch; writes selected coords.
// ---------------------------------------------------------------------------
__global__ __launch_bounds__(256) void fps_kernel(const float* __restrict__ xyz,
                                                  int N, int npoint,
                                                  float* __restrict__ new_xyz) {
  __shared__ float dist[2048];
  __shared__ float sval[256];
  __shared__ int   sidx[256];
  __shared__ int   sfar;
  const int b = blockIdx.x, tid = threadIdx.x;
  const float* P = xyz + (size_t)b * N * 3;
  for (int j = tid; j < N; j += 256) dist[j] = 1e10f;
  if (tid == 0) sfar = 0;
  __syncthreads();
  for (int i = 0; i < npoint; ++i) {
    const int far = sfar;
    const float cx = P[far * 3], cy = P[far * 3 + 1], cz = P[far * 3 + 2];
    if (tid == 0) {
      float* o = new_xyz + ((size_t)b * npoint + i) * 3;
      o[0] = cx; o[1] = cy; o[2] = cz;
    }
    float bv = -1.f; int bi = 0x7fffffff;
    for (int j = tid; j < N; j += 256) {
      const float dx = P[j * 3] - cx, dy = P[j * 3 + 1] - cy, dz = P[j * 3 + 2] - cz;
      const float nd = fminf(dist[j], dx * dx + dy * dy + dz * dz);
      dist[j] = nd;
      if (nd > bv) { bv = nd; bi = j; }
    }
    sval[tid] = bv; sidx[tid] = bi;
    __syncthreads();
    for (int off = 128; off > 0; off >>= 1) {
      if (tid < off) {
        const float ov = sval[tid + off]; const int oi = sidx[tid + off];
        if (ov > sval[tid] || (ov == sval[tid] && oi < sidx[tid])) {
          sval[tid] = ov; sidx[tid] = oi;
        }
      }
      __syncthreads();
    }
    if (tid == 0) sfar = sidx[0];
    __syncthreads();
  }
}

// ---------------------------------------------------------------------------
// Ball query: one wave per query; first nsample in-radius indices in ascending
// order (== reference's where/sort/slice), padded with the first hit.
// ---------------------------------------------------------------------------
__global__ __launch_bounds__(128) void ballq_kernel(const float* __restrict__ xyz,
                                                    const float* __restrict__ new_xyz,
                                                    int N, int S, float r2, int nsample,
                                                    int* __restrict__ out) {
  const int lane = threadIdx.x & 31;
  const int q = blockIdx.x * 4 + (threadIdx.x >> 5);
  const int b = q / S;
  const float* P = xyz + (size_t)b * N * 3;
  const float cx = new_xyz[q * 3], cy = new_xyz[q * 3 + 1], cz = new_xyz[q * 3 + 2];
  int* o = out + (size_t)q * nsample;
  int cnt = 0, first = 0;
  bool have_first = false;
  for (int base = 0; base < N && cnt < nsample; base += 32) {
    const int j = base + lane;
    bool pred = false;
    if (j < N) {
      const float dx = P[j * 3] - cx, dy = P[j * 3 + 1] - cy, dz = P[j * 3 + 2] - cz;
      pred = (dx * dx + dy * dy + dz * dz) <= r2;
    }
    const unsigned mask = (unsigned)__ballot(pred);
    if (!have_first && mask) { first = base + __builtin_ctz(mask); have_first = true; }
    const int rank = __popc(mask & ((1u << lane) - 1u));
    if (pred && cnt + rank < nsample) o[cnt + rank] = j;
    cnt += __popc(mask);
  }
  if (cnt > nsample) cnt = nsample;
  for (int k = cnt + lane; k < nsample; k += 32) o[k] = first;
}

// ---------------------------------------------------------------------------
// SA layer 1: groups of 32, channels 3(->pad32) -> 64 -> 64 -> 128, maxpool.
// One wave per group, 4 groups per block. Fully fused in LDS (bf16 acts).
// ---------------------------------------------------------------------------
__global__ __launch_bounds__(128) void sa1_mlp_kernel(
    const float* __restrict__ xyz, const float* __restrict__ new_xyz,
    const int* __restrict__ grp_idx,
    const bf16_t* __restrict__ w1, const bf16_t* __restrict__ w2,
    const bf16_t* __restrict__ w3, const float* __restrict__ bias,
    float* __restrict__ out) {
  __shared__ bf16_t X[4][32 * 32];
  __shared__ bf16_t H1[4][32 * 64];
  __shared__ bf16_t H2[4][32 * 64];
  const int w = threadIdx.x >> 5, lane = threadIdx.x & 31;
  const int gid = blockIdx.x * 4 + w;                 // b*512 + s
  const int b = gid >> 9;
  // ---- gather: lane m handles one neighbor
  {
    const int pi = grp_idx[(size_t)gid * 32 + lane];
    const float cx = new_xyz[gid * 3], cy = new_xyz[gid * 3 + 1], cz = new_xyz[gid * 3 + 2];
    const float* p = xyz + ((size_t)b * 2048 + pi) * 3;
    bf16_t* row = &X[w][lane * 32];
    row[0] = (bf16_t)(p[0] - cx);
    row[1] = (bf16_t)(p[1] - cy);
    row[2] = (bf16_t)(p[2] - cz);
#pragma unroll
    for (int k = 3; k < 32; ++k) row[k] = (bf16_t)0.f;
  }
  __syncthreads();
  // ---- conv1: Kpad=32, N=64
#pragma unroll
  for (int mt = 0; mt < 2; ++mt) {
    const v16bf a = load_a_tile(X[w], 32, mt * 16, 0);
#pragma unroll
    for (int nt = 0; nt < 4; ++nt) {
      v8f c = V8F_ZERO;
      c = wmma_bf16(a, load_b_tile(w1, 32, nt * 16, 0), c);
      store_act_tile(H1[w], 64, mt * 16, nt * 16, c, bias + 0);
    }
  }
  __syncthreads();
  // ---- conv2: K=64, N=64
#pragma unroll
  for (int mt = 0; mt < 2; ++mt) {
    const v16bf a0 = load_a_tile(H1[w], 64, mt * 16, 0);
    const v16bf a1 = load_a_tile(H1[w], 64, mt * 16, 32);
#pragma unroll
    for (int nt = 0; nt < 4; ++nt) {
      v8f c = V8F_ZERO;
      c = wmma_bf16(a0, load_b_tile(w2, 64, nt * 16, 0), c);
      c = wmma_bf16(a1, load_b_tile(w2, 64, nt * 16, 32), c);
      store_act_tile(H2[w], 64, mt * 16, nt * 16, c, bias + 64);
    }
  }
  __syncthreads();
  // ---- conv3: K=64, N=128, deferred bias+relu through maxpool
  const float* b3 = bias + 128;
#pragma unroll
  for (int nt = 0; nt < 8; ++nt) {
    float vmax = -1e30f;
#pragma unroll
    for (int mt = 0; mt < 2; ++mt) {
      v8f c = V8F_ZERO;
      c = wmma_bf16(load_a_tile(H2[w], 64, mt * 16, 0),  load_b_tile(w3, 64, nt * 16, 0),  c);
      c = wmma_bf16(load_a_tile(H2[w], 64, mt * 16, 32), load_b_tile(w3, 64, nt * 16, 32), c);
#pragma unroll
      for (int v = 0; v < 8; ++v) vmax = fmaxf(vmax, c[v]);
    }
    vmax = fmaxf(vmax, __shfl_xor(vmax, 16, 32));
    if (lane < 16) {
      const int n = nt * 16 + lane;
      const float x = vmax + b3[n];
      out[(size_t)gid * 128 + n] = x > 0.f ? x : 0.f;
    }
  }
}

// ---------------------------------------------------------------------------
// SA layer 2: groups of 64, channels 131(->pad160) -> 128 -> 128 -> 256.
// One block (4 waves) per group; waves split N-tiles.
// ---------------------------------------------------------------------------
__global__ __launch_bounds__(128) void sa2_mlp_kernel(
    const float* __restrict__ l1_xyz, const float* __restrict__ l1_feat,
    const float* __restrict__ new_xyz, const int* __restrict__ grp_idx,
    const bf16_t* __restrict__ w1, const bf16_t* __restrict__ w2,
    const bf16_t* __restrict__ w3, const float* __restrict__ bias,
    float* __restrict__ out) {
  __shared__ bf16_t X[64 * 160];
  __shared__ bf16_t H1[64 * 128];
  __shared__ bf16_t H2[64 * 128];
  const int tid = threadIdx.x, w = tid >> 5, lane = tid & 31;
  const int gid = blockIdx.x;                          // b*128 + s
  const int b = gid >> 7;
  // ---- gather: 64 rows, threads 0..63
  if (tid < 64) {
    const int pi = grp_idx[(size_t)gid * 64 + tid];
    const float cx = new_xyz[gid * 3], cy = new_xyz[gid * 3 + 1], cz = new_xyz[gid * 3 + 2];
    const float* p = l1_xyz + ((size_t)b * 512 + pi) * 3;
    bf16_t* row = &X[tid * 160];
    row[0] = (bf16_t)(p[0] - cx);
    row[1] = (bf16_t)(p[1] - cy);
    row[2] = (bf16_t)(p[2] - cz);
    const float* f = l1_feat + ((size_t)b * 512 + pi) * 128;
    for (int j = 0; j < 128; ++j) row[3 + j] = (bf16_t)f[j];
#pragma unroll
    for (int k = 131; k < 160; ++k) row[k] = (bf16_t)0.f;
  }
  __syncthreads();
  // ---- conv1: K=160 (5 steps), N=128 (8 tiles / 4 waves), M=64
  for (int mt = 0; mt < 4; ++mt) {
    v16bf a[5];
#pragma unroll
    for (int kt = 0; kt < 5; ++kt) a[kt] = load_a_tile(X, 160, mt * 16, kt * 32);
    for (int nt = w; nt < 8; nt += 4) {
      v8f c = V8F_ZERO;
#pragma unroll
      for (int kt = 0; kt < 5; ++kt)
        c = wmma_bf16(a[kt], load_b_tile(w1, 160, nt * 16, kt * 32), c);
      store_act_tile(H1, 128, mt * 16, nt * 16, c, bias + 0);
    }
  }
  __syncthreads();
  // ---- conv2: K=128 (4 steps), N=128
  for (int mt = 0; mt < 4; ++mt) {
    v16bf a[4];
#pragma unroll
    for (int kt = 0; kt < 4; ++kt) a[kt] = load_a_tile(H1, 128, mt * 16, kt * 32);
    for (int nt = w; nt < 8; nt += 4) {
      v8f c = V8F_ZERO;
#pragma unroll
      for (int kt = 0; kt < 4; ++kt)
        c = wmma_bf16(a[kt], load_b_tile(w2, 128, nt * 16, kt * 32), c);
      store_act_tile(H2, 128, mt * 16, nt * 16, c, bias + 128);
    }
  }
  __syncthreads();
  // ---- conv3: K=128, N=256 (16 tiles / 4 waves), maxpool over 64 rows
  float vmax[4] = {-1e30f, -1e30f, -1e30f, -1e30f};
  for (int mt = 0; mt < 4; ++mt) {
    v16bf a[4];
#pragma unroll
    for (int kt = 0; kt < 4; ++kt) a[kt] = load_a_tile(H2, 128, mt * 16, kt * 32);
#pragma unroll
    for (int j = 0; j < 4; ++j) {
      const int nt = w + 4 * j;
      v8f c = V8F_ZERO;
#pragma unroll
      for (int kt = 0; kt < 4; ++kt)
        c = wmma_bf16(a[kt], load_b_tile(w3, 128, nt * 16, kt * 32), c);
#pragma unroll
      for (int v = 0; v < 8; ++v) vmax[j] = fmaxf(vmax[j], c[v]);
    }
  }
  const float* b3 = bias + 256;
#pragma unroll
  for (int j = 0; j < 4; ++j) {
    float v = vmax[j];
    v = fmaxf(v, __shfl_xor(v, 16, 32));
    if (lane < 16) {
      const int n = (w + 4 * j) * 16 + lane;
      const float x = v + b3[n];
      out[(size_t)gid * 256 + n] = x > 0.f ? x : 0.f;
    }
  }
}

// ---------------------------------------------------------------------------
// SA layer 3 (group_all): 128 points, channels 259(->pad288)->256->512->1024,
// maxpool over the 128 points. gridDim = (32 batches, 4 splits of N3=1024).
// M tiled in 4 chunks of 32 rows; BUF reused for X (ld288) then H2 (ld512).
// ---------------------------------------------------------------------------
__global__ __launch_bounds__(128) void sa3_mlp_kernel(
    const float* __restrict__ l2_xyz, const float* __restrict__ l2_feat,
    const bf16_t* __restrict__ w1, const bf16_t* __restrict__ w2,
    const bf16_t* __restrict__ w3, const float* __restrict__ bias,
    float* __restrict__ out) {
  __shared__ bf16_t BUF[32 * 512];   // X (ld 288) then H2 (ld 512)
  __shared__ bf16_t H1[32 * 256];
  const int tid = threadIdx.x, w = tid >> 5, lane = tid & 31;
  const int b = blockIdx.x;
  const int ns = blockIdx.y;                           // conv3 channels [ns*256, +256)
  float vmax[4] = {-1e30f, -1e30f, -1e30f, -1e30f};
  for (int chunk = 0; chunk < 4; ++chunk) {
    __syncthreads();  // BUF reuse guard across chunks
    // ---- gather 32 rows (absolute coords + 256 feats), threads 0..31
    if (tid < 32) {
      const int r = chunk * 32 + tid;
      bf16_t* row = &BUF[tid * 288];
      const float* p = l2_xyz + ((size_t)b * 128 + r) * 3;
      row[0] = (bf16_t)p[0]; row[1] = (bf16_t)p[1]; row[2] = (bf16_t)p[2];
      const float* f = l2_feat + ((size_t)b * 128 + r) * 256;
      for (int j = 0; j < 256; ++j) row[3 + j] = (bf16_t)f[j];
#pragma unroll
      for (int k = 259; k < 288; ++k) row[k] = (bf16_t)0.f;
    }
    __syncthreads();
    // ---- conv1: K=288 (9 steps), N=256 (16 tiles / 4 waves), M=32
    for (int mt = 0; mt < 2; ++mt) {
      v16bf a[9];
#pragma unroll
      for (int kt = 0; kt < 9; ++kt) a[kt] = load_a_tile(BUF, 288, mt * 16, kt * 32);
      for (int nt = w; nt < 16; nt += 4) {
        v8f c = V8F_ZERO;
#pragma unroll
        for (int kt = 0; kt < 9; ++kt)
          c = wmma_bf16(a[kt], load_b_tile(w1, 288, nt * 16, kt * 32), c);
        store_act_tile(H1, 256, mt * 16, nt * 16, c, bias + 0);
      }
    }
    __syncthreads();
    // ---- conv2: K=256 (8 steps), N=512 (32 tiles / 4 waves), into BUF (ld 512)
    for (int mt = 0; mt < 2; ++mt) {
      v16bf a[8];
#pragma unroll
      for (int kt = 0; kt < 8; ++kt) a[kt] = load_a_tile(H1, 256, mt * 16, kt * 32);
      for (int nt = w; nt < 32; nt += 4) {
        v8f c = V8F_ZERO;
#pragma unroll
        for (int kt = 0; kt < 8; ++kt)
          c = wmma_bf16(a[kt], load_b_tile(w2, 256, nt * 16, kt * 32), c);
        store_act_tile(BUF, 512, mt * 16, nt * 16, c, bias + 256);
      }
    }
    __syncthreads();
    // ---- conv3: K=512 (16 steps), this block's 256 channels, maxpool partials
    for (int mt = 0; mt < 2; ++mt) {
      v8f acc[4] = {V8F_ZERO, V8F_ZERO, V8F_ZERO, V8F_ZERO};
      for (int kt = 0; kt < 16; ++kt) {
        const v16bf a = load_a_tile(BUF, 512, mt * 16, kt * 32);
#pragma unroll
        for (int j = 0; j < 4; ++j) {
          const int n0 = ns * 256 + (w * 4 + j) * 16;
          acc[j] = wmma_bf16(a, load_b_tile(w3, 512, n0, kt * 32), acc[j]);
        }
      }
#pragma unroll
      for (int j = 0; j < 4; ++j)
#pragma unroll
        for (int v = 0; v < 8; ++v) vmax[j] = fmaxf(vmax[j], acc[j][v]);
    }
  }
#pragma unroll
  for (int j = 0; j < 4; ++j) {
    float v = vmax[j];
    v = fmaxf(v, __shfl_xor(v, 16, 32));
    if (lane < 16) {
      const int n = ns * 256 + (w * 4 + j) * 16 + lane;
      const float x = v + bias[768 + n];
      out[(size_t)b * 1024 + n] = x > 0.f ? x : 0.f;
    }
  }
}

// ---------------------------------------------------------------------------
// Host orchestration
// ---------------------------------------------------------------------------
extern "C" void kernel_launch(void* const* d_in, const int* in_sizes, int n_in,
                              void* d_out, int out_size, void* d_ws, size_t ws_size,
                              hipStream_t stream) {
  (void)in_sizes; (void)n_in; (void)out_size; (void)ws_size;
  const float* xyz = (const float*)d_in[0];

  // ---- workspace carve (256B aligned)
  char* base = (char*)d_ws;
  size_t off = 0;
  auto carve = [&](size_t bytes) -> void* {
    void* p = base + off;
    off = (off + bytes + 255) & ~(size_t)255;
    return p;
  };
  float* l1_nx   = (float*)carve(32 * 512 * 3 * sizeof(float));
  int*   l1_idx  = (int*)  carve((size_t)32 * 512 * 32 * sizeof(int));
  float* l1_out  = (float*)carve((size_t)32 * 512 * 128 * sizeof(float));
  float* l2_nx   = (float*)carve(32 * 128 * 3 * sizeof(float));
  int*   l2_idx  = (int*)  carve((size_t)32 * 128 * 64 * sizeof(int));
  float* l2_out  = (float*)carve((size_t)32 * 128 * 256 * sizeof(float));
  bf16_t* wbuf   = (bf16_t*)carve((size_t)813056 * sizeof(bf16_t));
  float*  bbuf   = (float*) carve(2560 * sizeof(float));

  // ---- fold BN into bf16 weights + f32 biases (9 convs)
  static const int    CO[9]   = {64, 64, 128, 128, 128, 256, 256, 512, 1024};
  static const int    CI[9]   = {3, 64, 64, 131, 128, 128, 259, 256, 512};
  static const int    KP[9]   = {32, 64, 64, 160, 128, 128, 288, 256, 512};
  static const size_t WOFF[9] = {0, 2048, 6144, 14336, 34816, 51200, 83968, 157696, 288768};
  static const int    BOFF[9] = {0, 64, 128, 256, 384, 512, 768, 1024, 1536};
  for (int t = 0; t < 9; ++t) {
    const int a = 1 + t * 6;
    fold_kernel<<<CO[t], 64, 0, stream>>>(
        (const float*)d_in[a + 0], (const float*)d_in[a + 1], (const float*)d_in[a + 2],
        (const float*)d_in[a + 3], (const float*)d_in[a + 4], (const float*)d_in[a + 5],
        wbuf + WOFF[t], bbuf + BOFF[t], CI[t], KP[t]);
  }

  // ---- SA1
  fps_kernel<<<32, 256, 0, stream>>>(xyz, 2048, 512, l1_nx);
  ballq_kernel<<<(32 * 512) / 4, 128, 0, stream>>>(xyz, l1_nx, 2048, 512,
                                                   0.2f * 0.2f, 32, l1_idx);
  sa1_mlp_kernel<<<(32 * 512) / 4, 128, 0, stream>>>(
      xyz, l1_nx, l1_idx, wbuf + WOFF[0], wbuf + WOFF[1], wbuf + WOFF[2],
      bbuf + 0, l1_out);

  // ---- SA2
  fps_kernel<<<32, 256, 0, stream>>>(l1_nx, 512, 128, l2_nx);
  ballq_kernel<<<(32 * 128) / 4, 128, 0, stream>>>(l1_nx, l2_nx, 512, 128,
                                                   0.4f * 0.4f, 64, l2_idx);
  sa2_mlp_kernel<<<32 * 128, 128, 0, stream>>>(
      l1_nx, l1_out, l2_nx, l2_idx, wbuf + WOFF[3], wbuf + WOFF[4], wbuf + WOFF[5],
      bbuf + 256, l2_out);

  // ---- SA3 (group_all) -> d_out (32,1024)
  sa3_mlp_kernel<<<dim3(32, 4), 128, 0, stream>>>(
      l2_nx, l2_out, wbuf + WOFF[6], wbuf + WOFF[7], wbuf + WOFF[8],
      bbuf + 768, (float*)d_out);
}
